// MambaGUDU_58342835749272
// MI455X (gfx1250) — matrix operand order repaired and checked by reading, not compiled, not run on hardware
//
#include <hip/hip_runtime.h>
#include <hip/hip_bf16.h>
#include <math.h>
#include <stdint.h>

// ---------------------------------------------------------------------------
// CDNA5 (gfx1250) kernel set for the Mamba-DenseUNet reference.
//  - every GEMM-like op -> v_wmma_f32_16x16x32_bf16 (fp32 accumulate)
//  - implicit-GEMM conv3x3, fused epilogues, fully fused Mamba scan
//  - bf16-resident GEMM: double-buffered global_load_async_to_lds_b128
//  - TDM-staged GEMM variant: tensor_load_to_lds + s_wait_tensorcnt
// ---------------------------------------------------------------------------

typedef __bf16 bf16_t;
typedef __attribute__((ext_vector_type(4)))  bf16_t v4bf;
typedef __attribute__((ext_vector_type(8)))  bf16_t v8bf;
typedef __attribute__((ext_vector_type(16))) bf16_t v16bf;
typedef __attribute__((ext_vector_type(8)))  float  v8f;
typedef __attribute__((ext_vector_type(4)))  unsigned int u32x4;
typedef __attribute__((ext_vector_type(8)))  int i32x8;
typedef __attribute__((ext_vector_type(4)))  int i32x4;

#define GSTRIDE 40   // 32 K elements + 8 pad: every fragment/store 16B aligned

__device__ __forceinline__ float siluf(float x) { return x / (1.0f + __expf(-x)); }
__device__ __forceinline__ float sigmf(float x) { return 1.0f / (1.0f + __expf(-x)); }

// generic pointer -> 32-bit LDS offset (async / TDM destination address)
typedef __attribute__((address_space(3))) const void* lds_cptr_t;
__device__ __forceinline__ unsigned lds_off(const void* p) {
  return (unsigned)(uintptr_t)(lds_cptr_t)p;
}

// one 16-byte async copy global -> LDS (ASYNCcnt-tracked, no VGPR data)
__device__ __forceinline__ void async_b128(unsigned dst_lds, const void* src) {
  asm volatile("global_load_async_to_lds_b128 %0, %1, off"
               :: "v"(dst_lds), "v"(src) : "memory");
}
__device__ __forceinline__ void wait_async0() {
  asm volatile("s_wait_asynccnt 0" ::: "memory");
}

// ---------------------------------------------------------------------------
// Tensor Data Mover: 2-D bf16 tile load, global -> LDS, with HW padding that
// reproduces the GSTRIDE row pitch (pad 4 DWORDs after every 16 DWORDs).
// D# layout per CDNA5 ISA ch.8 (group0: flags/lds/global addr, group1: dims).
// ---------------------------------------------------------------------------
__device__ __forceinline__ void tdm_load_2d_bf16(
    unsigned lds_base, const void* gaddr,
    unsigned tensor_d0, unsigned tensor_d1,   // extents (elements) for OOB
    unsigned tile_d0, unsigned tile_d1,       // tile size (elements)
    unsigned stride_elems)                    // dim0 stride (elements)
{
#if defined(__has_builtin)
#if __has_builtin(__builtin_amdgcn_tensor_load_to_lds)
  unsigned long long ga = (unsigned long long)(uintptr_t)gaddr;
  u32x4 g0;
  g0[0] = 1u;                                        // count=1, user descriptor
  g0[1] = lds_base;                                  // lds_addr
  g0[2] = (unsigned)(ga & 0xffffffffu);              // global_addr[31:0]
  g0[3] = (unsigned)((ga >> 32) & 0x01ffffffu) | (2u << 30);  // addr hi | type=2
  i32x8 g1;
  // data_size=1 (2B) | pad_enable | pad_interval=16dw (3) | pad_amount=4dw (3)
  g1[0] = (int)((1u << 16) | (1u << 20) | (3u << 22) | (3u << 25));
  g1[1] = (int)((tensor_d0 & 0xffffu) << 16);                      // dim0[15:0]
  g1[2] = (int)(((tensor_d0 >> 16) & 0xffffu) | ((tensor_d1 & 0xffffu) << 16));
  g1[3] = (int)(((tensor_d1 >> 16) & 0xffffu) | ((tile_d0 & 0xffffu) << 16));
  g1[4] = (int)(tile_d1 & 0xffffu);                  // tile_dim1 | tile_dim2=0
  g1[5] = (int)stride_elems;                         // tensor_dim0_stride lo32
  g1[6] = 0;                                         // stride hi | dim1_stride
  g1[7] = 0;
  i32x4 gz = {0, 0, 0, 0};
#if __clang_major__ >= 23
  i32x8 gz8 = {0, 0, 0, 0, 0, 0, 0, 0};
  __builtin_amdgcn_tensor_load_to_lds(g0, g1, gz, gz, gz8, 0);
#else
  __builtin_amdgcn_tensor_load_to_lds(g0, g1, gz, gz, 0);
#endif
#endif
#endif
}

// ---------------------------------------------------------------------------
// Shared fragment-load + WMMA step (A/B tiles staged in LDS as bf16).
// ---------------------------------------------------------------------------
__device__ __forceinline__ void wmma_step(const bf16_t* As, const bf16_t* Bs,
                                          int lane, int wm, int wn,
                                          v8f& acc0, v8f& acc1) {
  const int arow = lane & 15;
  const int koff = (lane & 16) ? 8 : 0;
  const bf16_t* ap = &As[(wm + arow) * GSTRIDE];
  v8bf alo = *(const v8bf*)(ap + koff);
  v8bf ahi = *(const v8bf*)(ap + 16 + koff);
  v16bf af;
  #pragma unroll
  for (int i = 0; i < 8; ++i) { af[i] = alo[i]; af[i + 8] = ahi[i]; }

  const int bcol = lane & 15;
  const int kb   = (lane & 16) ? 8 : 0;
  const bf16_t* bp0 = &Bs[(wn + bcol) * GSTRIDE + kb];
  const bf16_t* bp1 = &Bs[(wn + 16 + bcol) * GSTRIDE + kb];
  v8bf b0lo = *(const v8bf*)bp0, b0hi = *(const v8bf*)(bp0 + 16);
  v8bf b1lo = *(const v8bf*)bp1, b1hi = *(const v8bf*)(bp1 + 16);
  v16bf bf0, bf1;
  #pragma unroll
  for (int i = 0; i < 8; ++i) {
    bf0[i] = b0lo[i]; bf0[i + 8] = b0hi[i];
    bf1[i] = b1lo[i]; bf1[i + 8] = b1hi[i];
  }
  acc0 = __builtin_amdgcn_wmma_f32_16x16x32_bf16(false, af, false, bf0,
                                                 (short)0, acc0, false, false);
  acc1 = __builtin_amdgcn_wmma_f32_16x16x32_bf16(false, af, false, bf1,
                                                 (short)0, acc1, false, false);
}

// vectorized fp32 row-tile stage (full tiles): float4 loads, packed bf16 stores
__device__ __forceinline__ void stage_rows_fast(bf16_t* S, const float* X,
                                                int ldx, int row0, int col0,
                                                int tid) {
  float4 v[2];
  int rr[2], cc[2];
  #pragma unroll
  for (int it = 0; it < 2; ++it) {
    int t = tid + it * 256;
    rr[it] = t >> 3;
    cc[it] = (t & 7) * 4;
    v[it] = *(const float4*)&X[(size_t)(row0 + rr[it]) * ldx + col0 + cc[it]];
  }
  #pragma unroll
  for (int it = 0; it < 2; ++it) {
    v4bf pk;
    pk[0] = (bf16_t)v[it].x; pk[1] = (bf16_t)v[it].y;
    pk[2] = (bf16_t)v[it].z; pk[3] = (bf16_t)v[it].w;
    *(v4bf*)&S[rr[it] * GSTRIDE + cc[it]] = pk;
  }
}

// guarded scalar row-tile stage (edge tiles): batch loads, then batch stores
__device__ __forceinline__ void stage_rows_slow(bf16_t* S, const float* X,
                                                int ldx, int row0, int col0,
                                                int nrow, int ncol, int tid) {
  float va[8];
  #pragma unroll
  for (int it = 0; it < 8; ++it) {
    int i = tid + it * 256;
    int r = i >> 5, c = i & 31;
    va[it] = (row0 + r < nrow && col0 + c < ncol)
                 ? X[(size_t)(row0 + r) * ldx + col0 + c] : 0.f;
  }
  #pragma unroll
  for (int it = 0; it < 8; ++it) {
    int i = tid + it * 256;
    int r = i >> 5, c = i & 31;
    S[r * GSTRIDE + c] = (bf16_t)va[it];
  }
}

// ---------------------------------------------------------------------------
// GEMM, B given transposed as BT[N,K]: C[M,N] = act(A[M,K] * BT^T + bias)
// ---------------------------------------------------------------------------
__global__ __launch_bounds__(256) void k_gemm_wmma_bt(
    const float* __restrict__ A, int lda,
    const float* __restrict__ BT, int ldbt,
    float* __restrict__ C, int ldc,
    const float* __restrict__ bias, int M, int N, int K, int act)
{
  __shared__ bf16_t As[64 * GSTRIDE];
  __shared__ bf16_t Bs[64 * GSTRIDE];
  const int tid = threadIdx.x, lane = tid & 31, wave = tid >> 5;
  const int m0 = blockIdx.y * 64, n0 = blockIdx.x * 64;
  const int wm = (wave & 3) * 16, wn = (wave >> 2) * 32;
  v8f acc0 = {}, acc1 = {};

  for (int k0 = 0; k0 < K; k0 += 32) {
    if (m0 + 64 <= M && k0 + 32 <= K) stage_rows_fast(As, A, lda, m0, k0, tid);
    else                              stage_rows_slow(As, A, lda, m0, k0, M, K, tid);
    if (n0 + 64 <= N && k0 + 32 <= K) stage_rows_fast(Bs, BT, ldbt, n0, k0, tid);
    else                              stage_rows_slow(Bs, BT, ldbt, n0, k0, N, K, tid);
    __syncthreads();
    wmma_step(As, Bs, lane, wm, wn, acc0, acc1);
    __syncthreads();
  }

  const int ccol = lane & 15, crow = (lane >> 4) * 8;
  #pragma unroll
  for (int r = 0; r < 8; ++r) {
    int gm = m0 + wm + crow + r;
    if (gm >= M) continue;
    int gn0 = n0 + wn + ccol, gn1 = gn0 + 16;
    float v0 = acc0[r], v1 = acc1[r];
    if (bias) {
      if (gn0 < N) v0 += bias[gn0];
      if (gn1 < N) v1 += bias[gn1];
    }
    if (act == 1)      { v0 = fmaxf(v0, 0.f); v1 = fmaxf(v1, 0.f); }
    else if (act == 2) { v0 = siluf(v0);      v1 = siluf(v1); }
    if (gn0 < N) C[(size_t)gm * ldc + gn0] = v0;
    if (gn1 < N) C[(size_t)gm * ldc + gn1] = v1;
  }
}

// ---------------------------------------------------------------------------
// GEMM with row-major B[K,N] (activation operand): B transposed into LDS.
// ---------------------------------------------------------------------------
__global__ __launch_bounds__(256) void k_gemm_wmma(
    const float* __restrict__ A, int lda,
    const float* __restrict__ B, int ldb,
    float* __restrict__ C, int ldc,
    const float* __restrict__ bias, int bias_mode,
    int M, int N, int K, int act)
{
  __shared__ bf16_t As[64 * GSTRIDE];
  __shared__ bf16_t Bs[64 * GSTRIDE];
  const int tid = threadIdx.x, lane = tid & 31, wave = tid >> 5;
  const int m0 = blockIdx.y * 64, n0 = blockIdx.x * 64;
  const int wm = (wave & 3) * 16, wn = (wave >> 2) * 32;
  v8f acc0 = {}, acc1 = {};

  for (int k0 = 0; k0 < K; k0 += 32) {
    if (m0 + 64 <= M && k0 + 32 <= K) stage_rows_fast(As, A, lda, m0, k0, tid);
    else                              stage_rows_slow(As, A, lda, m0, k0, M, K, tid);

    if (k0 + 32 <= K && n0 + 64 <= N) {
      float4 v[2]; int kk[2], nn[2];
      #pragma unroll
      for (int it = 0; it < 2; ++it) {
        int t = tid + it * 256;
        kk[it] = t >> 4;
        nn[it] = (t & 15) * 4;
        v[it] = *(const float4*)&B[(size_t)(k0 + kk[it]) * ldb + n0 + nn[it]];
      }
      #pragma unroll
      for (int it = 0; it < 2; ++it) {
        Bs[(nn[it] + 0) * GSTRIDE + kk[it]] = (bf16_t)v[it].x;
        Bs[(nn[it] + 1) * GSTRIDE + kk[it]] = (bf16_t)v[it].y;
        Bs[(nn[it] + 2) * GSTRIDE + kk[it]] = (bf16_t)v[it].z;
        Bs[(nn[it] + 3) * GSTRIDE + kk[it]] = (bf16_t)v[it].w;
      }
    } else {
      float vb[8];
      #pragma unroll
      for (int it = 0; it < 8; ++it) {
        int i = tid + it * 256;
        int c = i & 63, r = i >> 6;
        vb[it] = (k0 + r < K && n0 + c < N)
                     ? B[(size_t)(k0 + r) * ldb + n0 + c] : 0.f;
      }
      #pragma unroll
      for (int it = 0; it < 8; ++it) {
        int i = tid + it * 256;
        int c = i & 63, r = i >> 6;
        Bs[c * GSTRIDE + r] = (bf16_t)vb[it];
      }
    }
    __syncthreads();
    wmma_step(As, Bs, lane, wm, wn, acc0, acc1);
    __syncthreads();
  }

  const int ccol = lane & 15, crow = (lane >> 4) * 8;
  #pragma unroll
  for (int r = 0; r < 8; ++r) {
    int gm = m0 + wm + crow + r;
    if (gm >= M) continue;
    int gn0 = n0 + wn + ccol, gn1 = gn0 + 16;
    float v0 = acc0[r], v1 = acc1[r];
    if (bias) {
      if (bias_mode == 1) {
        if (gn0 < N) v0 += bias[gn0];
        if (gn1 < N) v1 += bias[gn1];
      } else if (bias_mode == 2) {
        float bv = bias[gm]; v0 += bv; v1 += bv;
      }
    }
    if (act == 1)      { v0 = fmaxf(v0, 0.f); v1 = fmaxf(v1, 0.f); }
    else if (act == 2) { v0 = siluf(v0);      v1 = siluf(v1); }
    if (gn0 < N) C[(size_t)gm * ldc + gn0] = v0;
    if (gn1 < N) C[(size_t)gm * ldc + gn1] = v1;
  }
}

// ---------------------------------------------------------------------------
// bf16-resident GEMM, double-buffered async global->LDS staging.
// Requires M % 64 == 0, N % 64 == 0, K % 32 == 0.
// Tile k+1 streams into the alternate LDS buffer while tile k is in the WMMAs.
// ---------------------------------------------------------------------------
__global__ __launch_bounds__(256) void k_gemm_bf16_async(
    const bf16_t* __restrict__ A,
    const bf16_t* __restrict__ BT,
    float* __restrict__ C, int ldc,
    int M, int N, int K, int act)
{
  __shared__ bf16_t As[2][64 * GSTRIDE];
  __shared__ bf16_t Bs[2][64 * GSTRIDE];
  const int tid = threadIdx.x, lane = tid & 31, wave = tid >> 5;
  const int m0 = blockIdx.y * 64, n0 = blockIdx.x * 64;
  const int wm = (wave & 3) * 16, wn = (wave >> 2) * 32;
  const int r = tid >> 2;            // row 0..63
  const int c = (tid & 3) * 8;       // 16B chunk 0,8,16,24
  v8f acc0 = {}, acc1 = {};

  auto issue = [&](int buf, int k0) {
    async_b128(lds_off(&As[buf][r * GSTRIDE + c]),
               &A[(size_t)(m0 + r) * K + k0 + c]);
    async_b128(lds_off(&Bs[buf][r * GSTRIDE + c]),
               &BT[(size_t)(n0 + r) * K + k0 + c]);
  };

  const int nk = K >> 5;
  issue(0, 0);
  for (int i = 0; i < nk; ++i) {
    const int cur = i & 1;
    wait_async0();
    __syncthreads();                       // buf[cur] visible to all waves
    if (i + 1 < nk) issue(cur ^ 1, (i + 1) << 5);
    wmma_step(As[cur], Bs[cur], lane, wm, wn, acc0, acc1);
    __syncthreads();                       // done reading buf[cur]
  }

  const int ccol = lane & 15, crow = (lane >> 4) * 8;
  #pragma unroll
  for (int rr = 0; rr < 8; ++rr) {
    int gm = m0 + wm + crow + rr;
    int gn0 = n0 + wn + ccol, gn1 = gn0 + 16;
    float v0 = acc0[rr], v1 = acc1[rr];
    if (act == 1)      { v0 = fmaxf(v0, 0.f); v1 = fmaxf(v1, 0.f); }
    else if (act == 2) { v0 = siluf(v0);      v1 = siluf(v1); }
    C[(size_t)gm * ldc + gn0] = v0;
    C[(size_t)gm * ldc + gn1] = v1;
  }
}

// ---------------------------------------------------------------------------
// bf16-resident GEMM with the B tile staged by the Tensor Data Mover.
// One wave issues a single D# per K-step (EXEC-independent DMA of a 64x32
// bf16 tile with HW row padding); A is staged with per-lane async copies.
// Requires M % 64 == 0, N % 64 == 0, K % 32 == 0.
// ---------------------------------------------------------------------------
__global__ __launch_bounds__(256) void k_gemm_bf16_tdm(
    const bf16_t* __restrict__ A,
    const bf16_t* __restrict__ BT,
    float* __restrict__ C, int ldc,
    int M, int N, int K, int act)
{
  __shared__ bf16_t As[64 * GSTRIDE];
  __shared__ bf16_t Bs[64 * GSTRIDE];
  const int tid = threadIdx.x, lane = tid & 31, wave = tid >> 5;
  const int m0 = blockIdx.y * 64, n0 = blockIdx.x * 64;
  const int wm = (wave & 3) * 16, wn = (wave >> 2) * 32;
  const int r = tid >> 2;
  const int c = (tid & 3) * 8;
  v8f acc0 = {}, acc1 = {};

  for (int k0 = 0; k0 < K; k0 += 32) {
    async_b128(lds_off(&As[r * GSTRIDE + c]),
               &A[(size_t)(m0 + r) * K + k0 + c]);
    if (wave == 0) {
      tdm_load_2d_bf16(lds_off(&Bs[0]), &BT[(size_t)n0 * K + k0],
                       (unsigned)(K - k0), (unsigned)(N - n0),
                       32u, 64u, (unsigned)K);
#if defined(__has_builtin)
#if __has_builtin(__builtin_amdgcn_s_wait_tensorcnt)
      __builtin_amdgcn_s_wait_tensorcnt(0);
#endif
#endif
    }
    wait_async0();
    __syncthreads();
    wmma_step(As, Bs, lane, wm, wn, acc0, acc1);
    __syncthreads();
  }

  const int ccol = lane & 15, crow = (lane >> 4) * 8;
  #pragma unroll
  for (int rr = 0; rr < 8; ++rr) {
    int gm = m0 + wm + crow + rr;
    int gn0 = n0 + wn + ccol, gn1 = gn0 + 16;
    float v0 = acc0[rr], v1 = acc1[rr];
    if (act == 1)      { v0 = fmaxf(v0, 0.f); v1 = fmaxf(v1, 0.f); }
    else if (act == 2) { v0 = siluf(v0);      v1 = siluf(v1); }
    C[(size_t)gm * ldc + gn0] = v0;
    C[(size_t)gm * ldc + gn1] = v1;
  }
}

// ---------------------------------------------------------------------------
// Implicit-GEMM 3x3 conv (NCHW, pad=1): M=H*W, N=Cout, K=Cin*9.
// ---------------------------------------------------------------------------
__global__ __launch_bounds__(256) void k_conv3x3_wmma(
    const float* __restrict__ X,
    const float* __restrict__ Wt,   // [Cout, Cin*9]  (natural [N][K])
    const float* __restrict__ bias,
    float* __restrict__ Y,          // [Cout, H*W]
    int Cin, int Cout, int H, int W, int act)
{
  __shared__ bf16_t As[64 * GSTRIDE];
  __shared__ bf16_t Bs[64 * GSTRIDE];
  const int M = H * W, N = Cout, K = Cin * 9;
  const int tid = threadIdx.x, lane = tid & 31, wave = tid >> 5;
  const int m0 = blockIdx.y * 64, n0 = blockIdx.x * 64;
  const int wm = (wave & 3) * 16, wn = (wave >> 2) * 32;
  v8f acc0 = {}, acc1 = {};

  for (int k0 = 0; k0 < K; k0 += 32) {
    float va[8];
    #pragma unroll
    for (int it = 0; it < 8; ++it) {
      int i = tid + it * 256;
      int rr = i >> 5, cc = i & 31;
      int p = m0 + rr, gk = k0 + cc;
      float v = 0.f;
      if (p < M && gk < K) {
        int ci = gk / 9, t = gk - ci * 9;
        int dy = t / 3 - 1;
        int dx = t - (t / 3) * 3 - 1;
        int py = p / W + dy;
        int px = p - (p / W) * W + dx;
        if (py >= 0 && py < H && px >= 0 && px < W)
          v = X[((size_t)ci * H + py) * W + px];
      }
      va[it] = v;
    }
    #pragma unroll
    for (int it = 0; it < 8; ++it) {
      int i = tid + it * 256;
      As[(i >> 5) * GSTRIDE + (i & 31)] = (bf16_t)va[it];
    }
    if (n0 + 64 <= N && k0 + 32 <= K) stage_rows_fast(Bs, Wt, K, n0, k0, tid);
    else                              stage_rows_slow(Bs, Wt, K, n0, k0, N, K, tid);
    __syncthreads();
    wmma_step(As, Bs, lane, wm, wn, acc0, acc1);
    __syncthreads();
  }

  const int ccol = lane & 15, crow = (lane >> 4) * 8;
  #pragma unroll
  for (int rr = 0; rr < 8; ++rr) {
    int gm = m0 + wm + crow + rr;
    if (gm >= M) continue;
    int gn0 = n0 + wn + ccol, gn1 = gn0 + 16;
    float v0 = acc0[rr], v1 = acc1[rr];
    if (bias) {
      if (gn0 < N) v0 += bias[gn0];
      if (gn1 < N) v1 += bias[gn1];
    }
    if (act == 1) { v0 = fmaxf(v0, 0.f); v1 = fmaxf(v1, 0.f); }
    if (gn0 < N) Y[(size_t)gn0 * M + gm] = v0;
    if (gn1 < N) Y[(size_t)gn1 * M + gm] = v1;
  }
}

// ---------------------------------------------------------------------------
// Fully fused Mamba selective scan (one thread per inner channel).
// ---------------------------------------------------------------------------
__global__ void k_mamba_scan(
    const float* __restrict__ dtraw, int ldt,
    const float* __restrict__ dt_bias,
    const float* __restrict__ xm, int ldx,
    const float* __restrict__ Bm, int ldB,
    const float* __restrict__ Cm, int ldC,
    const float* __restrict__ A_log,
    const float* __restrict__ Dp,
    float* __restrict__ y, int ldy,
    int L, int di)
{
  int d = blockIdx.x * blockDim.x + threadIdx.x;
  if (d >= di) return;
  float A[8], h[8];
  #pragma unroll
  for (int s = 0; s < 8; ++s) { A[s] = -__expf(A_log[(size_t)d * 8 + s]); h[s] = 0.f; }
  const float bias = dt_bias[d];
  const float Dd   = Dp[d];
  for (int l = 0; l < L; ++l) {
    float dt = dtraw[(size_t)l * ldt + d] + bias;
    dt = (dt > 20.f) ? dt : log1pf(__expf(dt));
    float xv = xm[(size_t)l * ldx + d];
    float dtx = dt * xv;
    float acc = 0.f;
    #pragma unroll
    for (int s = 0; s < 8; ++s) {
      float dA = __expf(dt * A[s]);
      h[s] = dA * h[s] + dtx * Bm[(size_t)l * ldB + s];
      acc += h[s] * Cm[(size_t)l * ldC + s];
    }
    y[(size_t)l * ldy + d] = acc + xv * Dd;
  }
}

// ---------------------------------------------------------------------------
// Elementwise / reduction / cast helpers
// ---------------------------------------------------------------------------
__global__ void k_fill(float* __restrict__ p, size_t n, float scale) {
  size_t i = blockIdx.x * (size_t)blockDim.x + threadIdx.x;
  if (i < n) p[i] = scale * (float)((int)(i % 251u) - 125);
}

__global__ void k_cast_bf16(const float* __restrict__ x, bf16_t* __restrict__ y,
                            size_t n) {
  size_t i = blockIdx.x * (size_t)blockDim.x + threadIdx.x;
  if (i < n) y[i] = (bf16_t)x[i];
}

__global__ void k_bn_relu(const float* __restrict__ x, const float* __restrict__ sc,
                          const float* __restrict__ bi, float* __restrict__ y,
                          int C, int HW) {
  size_t idx = blockIdx.x * (size_t)blockDim.x + threadIdx.x;
  if (idx >= (size_t)C * HW) return;
  int c = (int)(idx / HW);
  y[idx] = fmaxf(x[idx] * sc[c] + bi[c], 0.f);
}

__global__ void k_chan_pool(const float* __restrict__ x, float* __restrict__ avg,
                            float* __restrict__ mx, int HW) {
  __shared__ float rs[8], rm[8];
  int c = blockIdx.x;
  float s = 0.f, m = -1e30f;
  for (int p = threadIdx.x; p < HW; p += blockDim.x) {
    float v = x[(size_t)c * HW + p];
    s += v; m = fmaxf(m, v);
  }
  for (int off = 16; off > 0; off >>= 1) {
    s += __shfl_down(s, off, 32);
    m = fmaxf(m, __shfl_down(m, off, 32));
  }
  int lane = threadIdx.x & 31, wv = threadIdx.x >> 5;
  if (lane == 0) { rs[wv] = s; rm[wv] = m; }
  __syncthreads();
  if (threadIdx.x == 0) {
    int nw = (blockDim.x + 31) >> 5;
    float ts = 0.f, tm = -1e30f;
    for (int i = 0; i < nw; ++i) { ts += rs[i]; tm = fmaxf(tm, rm[i]); }
    avg[c] = ts / (float)HW;
    mx[c]  = tm;
  }
}

__global__ void k_fc_vec(const float* __restrict__ v, const float* __restrict__ Wm,
                         float* __restrict__ o, int In, int Out, int act) {
  int j = blockIdx.x * blockDim.x + threadIdx.x;
  if (j >= Out) return;
  float a = 0.f;
  for (int i = 0; i < In; ++i) a += v[i] * Wm[(size_t)i * Out + j];
  if (act == 1) a = fmaxf(a, 0.f);
  o[j] = a;
}

__global__ void k_rmsnorm(const float* __restrict__ x, const float* __restrict__ wn,
                          float* __restrict__ y, int C) {
  __shared__ float red[8];
  int l = blockIdx.x;
  float s = 0.f;
  for (int c = threadIdx.x; c < C; c += blockDim.x) {
    float v = x[(size_t)l * C + c]; s += v * v;
  }
  for (int off = 16; off > 0; off >>= 1) s += __shfl_down(s, off, 32);
  int lane = threadIdx.x & 31, wv = threadIdx.x >> 5;
  if (lane == 0) red[wv] = s;
  __syncthreads();
  if (threadIdx.x == 0) {
    int nw = (blockDim.x + 31) >> 5;
    float t = 0.f;
    for (int i = 0; i < nw; ++i) t += red[i];
    red[0] = rsqrtf(t / (float)C + 1e-6f);
  }
  __syncthreads();
  float sc = red[0];
  for (int c = threadIdx.x; c < C; c += blockDim.x)
    y[(size_t)l * C + c] = x[(size_t)l * C + c] * sc * wn[c];
}

__global__ void k_dwconv1d_silu(const float* __restrict__ xz, int inld,
                                const float* __restrict__ cw, const float* __restrict__ cb,
                                float* __restrict__ xc, int L, int di) {
  int idx = blockIdx.x * blockDim.x + threadIdx.x;
  if (idx >= L * di) return;
  int l = idx / di, d = idx - (idx / di) * di;
  float acc = cb[d];
  #pragma unroll
  for (int j = 0; j < 4; ++j) {
    int ll = l - 3 + j;
    if (ll >= 0) acc += cw[d * 4 + j] * xz[(size_t)ll * inld + d];
  }
  xc[idx] = siluf(acc);
}

__global__ void k_silu_gate(float* __restrict__ y, const float* __restrict__ z,
                            int zld, int L, int di) {
  int idx = blockIdx.x * blockDim.x + threadIdx.x;
  if (idx >= L * di) return;
  int l = idx / di, d = idx - (idx / di) * di;
  y[idx] *= siluf(z[(size_t)l * zld + d]);
}

__global__ void k_seq_gather(const float* __restrict__ x, float* __restrict__ seq,
                             int C, int HW, int L) {
  int idx = blockIdx.x * blockDim.x + threadIdx.x;
  if (idx >= L * C) return;
  int l = idx / C, c = idx - (idx / C) * C;
  seq[idx] = x[(size_t)c * HW + l];
}

__global__ void k_seq_scatter(const float* __restrict__ seq, float* __restrict__ x,
                              int C, int HW, int L) {
  int idx = blockIdx.x * blockDim.x + threadIdx.x;
  if (idx >= L * C) return;
  int l = idx / C, c = idx - (idx / C) * C;
  x[(size_t)c * HW + l] = seq[idx];
}

__global__ void k_attn_apply(const float* __restrict__ x, const float* __restrict__ a1v,
                             const float* __restrict__ a2v, const float* __restrict__ mt,
                             float* __restrict__ y, int C, int HW) {
  size_t idx = blockIdx.x * (size_t)blockDim.x + threadIdx.x;
  if (idx >= (size_t)C * HW) return;
  int c = (int)(idx / HW);
  y[idx] = x[idx] * sigmf(a1v[c] + a2v[c]) + 0.1f * mt[idx];
}

// ---------------------------------------------------------------------------
// Orchestration
// ---------------------------------------------------------------------------
static inline dim3 gemm_grid(int M, int N) { return dim3((N + 63) / 64, (M + 63) / 64); }

extern "C" void kernel_launch(void* const* d_in, const int* in_sizes, int n_in,
                              void* d_out, int out_size, void* d_ws, size_t ws_size,
                              hipStream_t stream) {
  (void)in_sizes; (void)n_in;
  const float* x   = (const float*)d_in[0];   // [1,1,96,96]
  float*       out = (float*)d_out;           // [4, 96*96]

  const int H = 96, Wd = 96, HW = H * Wd;
  const int C0 = 64, G = 32;
  const int L = 576;
  const int Dm = 64, Di = 128, R = 16;

  size_t cur = 0;
  auto alloc = [&](size_t n) -> float* {
    float* p = (float*)((char*)d_ws + cur);
    cur += n * sizeof(float);
    cur = (cur + 255) & ~(size_t)255;
    return p;
  };
  float* w0   = alloc(C0 * 1 * 9);  float* b0  = alloc(C0);
  float* bns  = alloc(C0);          float* bnb = alloc(C0);
  float* w1   = alloc(G * C0 * 9);  float* b1  = alloc(G);
  float* fc1  = alloc(C0 * 4);      float* fc2 = alloc(4 * C0);
  float* nw   = alloc(Dm);
  float* Win  = alloc((size_t)(2 * Di) * Dm);    // [N=256, K=64]
  float* cw   = alloc(Di * 4);      float* cb  = alloc(Di);
  float* Wx   = alloc((size_t)(R + 16) * Di);    // [N=32, K=128]
  float* Wdt  = alloc((size_t)Di * R);           // [N=128, K=16]
  float* dtb  = alloc(Di);
  float* Alog = alloc(Di * 8);      float* Dp  = alloc(Di);
  float* Wout = alloc((size_t)Dm * Di);          // [N=64, K=128]
  float* hwt  = alloc(4 * C0);      float* hb  = alloc(4);
  float* a0   = alloc((size_t)C0 * HW);
  float* a1   = alloc((size_t)C0 * HW);
  float* a2g  = alloc((size_t)G * HW);
  float* a3   = alloc((size_t)C0 * HW);
  float* mt   = alloc((size_t)C0 * HW);
  float* avgv = alloc(C0);  float* mxv = alloc(C0);
  float* t1   = alloc(4);   float* t2  = alloc(4);
  float* at1  = alloc(C0);  float* at2 = alloc(C0);
  float* seq  = alloc((size_t)L * Dm);
  float* xn   = alloc((size_t)L * Dm);
  float* xz   = alloc((size_t)L * 2 * Di);
  float* xc   = alloc((size_t)L * Di);
  float* dbc  = alloc((size_t)L * (R + 16));
  float* dtr  = alloc((size_t)L * Di);
  float* yb   = alloc((size_t)L * Di);
  float* mo   = alloc((size_t)L * Dm);
  bf16_t* xnb   = (bf16_t*)alloc((size_t)L * Dm / 2);
  bf16_t* WinT  = (bf16_t*)alloc((size_t)(2 * Di) * Dm / 2);
  bf16_t* ybb   = (bf16_t*)alloc((size_t)L * Di / 2);
  bf16_t* WoutT = (bf16_t*)alloc((size_t)Dm * Di / 2);
  size_t total_floats = cur / sizeof(float);

  if (cur > ws_size) {
    size_t n = (size_t)out_size;
    k_fill<<<dim3((unsigned)((n + 255) / 256)), 256, 0, stream>>>(out, n, 0.f);
    return;
  }

  // 0) deterministic pseudo-weight fill of entire workspace
  k_fill<<<dim3((unsigned)((total_floats + 255) / 256)), 256, 0, stream>>>(
      (float*)d_ws, total_floats, 0.01f);

  // 1) init conv 3x3: 1 -> 64 (fused relu)
  k_conv3x3_wmma<<<gemm_grid(HW, C0), 256, 0, stream>>>(
      x, w0, b0, a0, 1, C0, H, Wd, 1);

  // 2) bn + relu
  k_bn_relu<<<dim3((unsigned)(((size_t)C0 * HW + 255) / 256)), 256, 0, stream>>>(
      a0, bns, bnb, a1, C0, HW);

  // 3) dense-block growth conv 3x3: 64 -> 32 (fused relu)
  k_conv3x3_wmma<<<gemm_grid(HW, G), 256, 0, stream>>>(
      a1, w1, b1, a2g, C0, G, H, Wd, 1);

  // 4) channel attention
  k_chan_pool<<<C0, 256, 0, stream>>>(a1, avgv, mxv, HW);
  k_fc_vec<<<1, 64, 0, stream>>>(avgv, fc1, t1, C0, 4, 1);
  k_fc_vec<<<1, 64, 0, stream>>>(t1, fc2, at1, 4, C0, 0);
  k_fc_vec<<<1, 64, 0, stream>>>(mxv, fc1, t2, C0, 4, 1);
  k_fc_vec<<<1, 64, 0, stream>>>(t2, fc2, at2, 4, C0, 0);

  // 5) mamba block, L=576, d=64, di=128
  k_seq_gather<<<dim3((L * Dm + 255) / 256), 256, 0, stream>>>(a1, seq, Dm, HW, L);
  k_rmsnorm<<<L, 256, 0, stream>>>(seq, nw, xn, Dm);
  // async double-buffered GEMM: xz = xn @ W_in  [576,64]x[64,256]
  k_cast_bf16<<<dim3((L * Dm + 255) / 256), 256, 0, stream>>>(xn, xnb, (size_t)L * Dm);
  k_cast_bf16<<<dim3((2 * Di * Dm + 255) / 256), 256, 0, stream>>>(
      Win, WinT, (size_t)(2 * Di) * Dm);
  k_gemm_bf16_async<<<gemm_grid(L, 2 * Di), 256, 0, stream>>>(
      xnb, WinT, xz, 2 * Di, L, 2 * Di, Dm, 0);
  // depthwise causal conv1d + silu
  k_dwconv1d_silu<<<dim3((L * Di + 255) / 256), 256, 0, stream>>>(
      xz, 2 * Di, cw, cb, xc, L, Di);
  // dbc = xm @ W_x  [576,128]x[128,32]
  k_gemm_wmma_bt<<<gemm_grid(L, R + 16), 256, 0, stream>>>(
      xc, Di, Wx, Di, dbc, R + 16, nullptr, L, R + 16, Di, 0);
  // dt_raw = dbc[:, :R] @ W_dt  [576,16]x[16,128]
  k_gemm_wmma_bt<<<gemm_grid(L, Di), 256, 0, stream>>>(
      dbc, R + 16, Wdt, R, dtr, Di, nullptr, L, Di, R, 0);
  // fused selective scan
  k_mamba_scan<<<dim3((Di + 127) / 128), 128, 0, stream>>>(
      dtr, Di, dtb, xc, Di, dbc + R, R + 16, dbc + R + 8, R + 16,
      Alog, Dp, yb, Di, L, Di);
  // y *= silu(z)
  k_silu_gate<<<dim3((L * Di + 255) / 256), 256, 0, stream>>>(
      yb, xz + Di, 2 * Di, L, Di);
  // mo = y @ W_out  [576,128]x[128,64]  (TDM-staged B tile)
  k_cast_bf16<<<dim3((L * Di + 255) / 256), 256, 0, stream>>>(yb, ybb, (size_t)L * Di);
  k_cast_bf16<<<dim3((Dm * Di + 255) / 256), 256, 0, stream>>>(
      Wout, WoutT, (size_t)Dm * Di);
  k_gemm_bf16_tdm<<<gemm_grid(L, Dm), 256, 0, stream>>>(
      ybb, WoutT, mo, Dm, L, Dm, Di, 0);

  // 6) scatter mamba output back, fused channel-attn apply
  k_fill<<<dim3((unsigned)(((size_t)C0 * HW + 255) / 256)), 256, 0, stream>>>(
      mt, (size_t)C0 * HW, 0.f);
  k_seq_scatter<<<dim3((L * Dm + 255) / 256), 256, 0, stream>>>(mo, mt, Dm, HW, L);
  k_attn_apply<<<dim3((unsigned)(((size_t)C0 * HW + 255) / 256)), 256, 0, stream>>>(
      a1, at1, at2, mt, a3, C0, HW);

  // 7) head: 1x1 conv as GEMM  [4,64] x [64,HW] -> out [4,HW]
  k_gemm_wmma<<<gemm_grid(4, HW), 256, 0, stream>>>(
      hwt, C0, a3, HW, out, HW, hb, 2, 4, HW, C0, 0);
  (void)out_size; (void)a2g;
}
